// SumLayer_82935818486453
// MI455X (gfx1250) — matrix operand order repaired and checked by reading, not compiled, not run on hardware
//
#include <hip/hip_runtime.h>
#include <math.h>
#include <stdint.h>

typedef __attribute__((ext_vector_type(16))) _Float16     v16h;
typedef __attribute__((ext_vector_type(8)))  float        v8f;
typedef __attribute__((ext_vector_type(4)))  unsigned int v4u;
typedef __attribute__((ext_vector_type(8)))  int          v8i;
typedef __attribute__((ext_vector_type(4)))  int          v4i;

#define BB   256
#define NN   8192
#define CC   32768
#define NSE  524288
#define NT   (NN / 16)              // 512 row tiles
#define PADMAX (NSE + NT * 64)      // worst-case padded edge stream (557056)

#if __has_builtin(__builtin_amdgcn_tensor_load_to_lds) && \
    __has_builtin(__builtin_amdgcn_s_wait_tensorcnt)
#define USE_TDM 1
#else
#define USE_TDM 0
#endif

// ---------------------------------------------------------------------------
// Kernel 1: pT16[c*B + b] = (f16) exp(child_ll[b*C + c])   (LDS-tiled transpose)
// ---------------------------------------------------------------------------
__global__ void exp_transpose_kernel(const float* __restrict__ child_ll,
                                     _Float16* __restrict__ pT) {
    __shared__ _Float16 tile[32][33];
    const int c0 = blockIdx.x * 32;
    const int b0 = blockIdx.y * 32;
    const int tx = threadIdx.x;     // 0..31
    const int ty = threadIdx.y;     // 0..7
#pragma unroll
    for (int i = 0; i < 32; i += 8) {
        tile[ty + i][tx] = (_Float16)__expf(child_ll[(size_t)(b0 + ty + i) * CC + (c0 + tx)]);
    }
    __syncthreads();
#pragma unroll
    for (int i = 0; i < 32; i += 8) {
        pT[(size_t)(c0 + ty + i) * BB + (b0 + tx)] = tile[tx][ty + i];
    }
}

// ---------------------------------------------------------------------------
// Kernel 2: CSR row offsets from sorted rows[]
// ---------------------------------------------------------------------------
__global__ void build_row_start_kernel(const int* __restrict__ rows,
                                       int* __restrict__ row_start) {
    const int e = blockIdx.x * blockDim.x + threadIdx.x;
    if (e >= NSE) return;
    const int r = rows[e];
    if (e == 0) {
        for (int rr = 0; rr <= r; ++rr) row_start[rr] = 0;
    } else {
        const int rp = rows[e - 1];
        for (int rr = rp + 1; rr <= r; ++rr) row_start[rr] = e;
    }
    if (e == NSE - 1) {
        for (int rr = r + 1; rr <= NN; ++rr) row_start[rr] = NSE;
    }
}

// ---------------------------------------------------------------------------
// Kernel 3: per-tile padded offsets (pad each tile's edge count to 64n).
// One block of NT threads, Hillis-Steele inclusive scan.
// ---------------------------------------------------------------------------
__global__ void tile_scan_kernel(const int* __restrict__ row_start,
                                 int* __restrict__ tile_pstart) {
    __shared__ int s[NT];
    const int t = threadIdx.x;                 // 0..NT-1
    const int cnt = row_start[t * 16 + 16] - row_start[t * 16];
    s[t] = (cnt + 63) & ~63;
    __syncthreads();
    for (int d = 1; d < NT; d <<= 1) {
        const int v = (t >= d) ? s[t - d] : 0;
        __syncthreads();
        s[t] += v;
        __syncthreads();
    }
    if (t == 0) tile_pstart[0] = 0;
    tile_pstart[t + 1] = s[t];
}

// ---------------------------------------------------------------------------
// Kernel 4: zero-fill padded streams (padding edges: colB=0, rw=0 -> w=0)
// ---------------------------------------------------------------------------
__global__ void fill_pad_kernel(int* __restrict__ colBp, unsigned* __restrict__ rwp) {
    const int i = blockIdx.x * blockDim.x + threadIdx.x;
    if (i < PADMAX) { colBp[i] = 0; rwp[i] = 0u; }
}

// ---------------------------------------------------------------------------
// Kernel 5: scatter pre-digested edges into padded per-tile streams:
//   colBp[dst] = cols[e]*B*2   (BYTE offset into pT)
//   rwp[dst]   = f16bits(exp(lw[e]))<<16 | (rows[e] & 15)
// ---------------------------------------------------------------------------
__global__ void scatter_edges_kernel(const float* __restrict__ lw,
                                     const int* __restrict__ rows,
                                     const int* __restrict__ cols,
                                     const int* __restrict__ row_start,
                                     const int* __restrict__ tile_pstart,
                                     int* __restrict__ colBp,
                                     unsigned* __restrict__ rwp) {
    const int e = blockIdx.x * blockDim.x + threadIdx.x;
    if (e >= NSE) return;
    const int r = rows[e];
    const int t = r >> 4;
    const int dst = tile_pstart[t] + (e - row_start[t << 4]);
    colBp[dst] = cols[e] * (BB * 2);
    const _Float16 hw = (_Float16)__expf(lw[e]);
    const unsigned short wb = __builtin_bit_cast(unsigned short, hw);
    rwp[dst] = ((unsigned)wb << 16) | (unsigned)(r & 15);
}

// ---------------------------------------------------------------------------
// Kernel 6: z[r] = logsumexp(log_w over row r)   (one wave32 per row)
// ---------------------------------------------------------------------------
__global__ void compute_z_kernel(const float* __restrict__ lw,
                                 const int* __restrict__ row_start,
                                 float* __restrict__ z) {
    const int wave = threadIdx.x >> 5;
    const int lane = threadIdx.x & 31;
    const int r    = blockIdx.x * 8 + wave;
    const int s    = row_start[r];
    const int e1   = row_start[r + 1];
    float mx = -INFINITY;
    for (int e = s + lane; e < e1; e += 32) mx = fmaxf(mx, lw[e]);
#pragma unroll
    for (int off = 16; off > 0; off >>= 1) mx = fmaxf(mx, __shfl_xor(mx, off, 32));
    float sum = 0.0f;
    for (int e = s + lane; e < e1; e += 32) sum += __expf(lw[e] - mx);
#pragma unroll
    for (int off = 16; off > 0; off >>= 1) sum += __shfl_xor(sum, off, 32);
    if (lane == 0) z[r] = mx + logf(sum);
}

// ---------------------------------------------------------------------------
// TDM helper: 1-line tensor load of 64 dwords to LDS (D# per ISA §8.3/8.4).
// clang-23 (this toolchain): 6-arg builtin; clang-22 (ROCm 7.2): 5-arg.
// ---------------------------------------------------------------------------
#if USE_TDM
__device__ __forceinline__ void tdm_load_64dw(unsigned lds_off, const void* gsrc,
                                              unsigned rem_dwords) {
    const unsigned long long ga = (unsigned long long)(uintptr_t)gsrc;
    v4u g0;
    g0.x = 1u;                                            // count=1, user mode
    g0.y = lds_off;                                       // lds_addr
    g0.z = (unsigned)(ga & 0xffffffffu);                  // global_addr[31:0]
    g0.w = (unsigned)((ga >> 32) & 0x01ffffffu) | 0x80000000u; // ga[56:32]|type=2
    v8i g1;
    g1[0] = 0x00020000;                                   // data_size = 4 bytes
    g1[1] = (int)((rem_dwords & 0xffffu) << 16);          // tensor_dim0[15:0]
    g1[2] = (int)((rem_dwords >> 16) | (1u << 16));       // dim0[31:16]|tensor_dim1=1
    g1[3] = (int)(64u << 16);                             // tile_dim0 = 64
    g1[4] = 0;                                            // tile_dim1/2 unused
    g1[5] = (int)rem_dwords;                              // tensor_dim0_stride[31:0]
    g1[6] = 0;
    g1[7] = 0;
    const v4i z4 = {0, 0, 0, 0};
#if __clang_major__ >= 23
    const v8i z8 = {0, 0, 0, 0, 0, 0, 0, 0};
    __builtin_amdgcn_tensor_load_to_lds(g0, g1, z4, z4, z8, 0);
#else
    __builtin_amdgcn_tensor_load_to_lds(g0, g1, z4, z4, 0);
#endif
}
#endif

// ---------------------------------------------------------------------------
// Kernel 7: WMMA SpMM over padded per-tile streams. Block = 8 waves, one
// 16-node tile per block; wave w owns TWO interleaved batch tiles
// (even: 32w+2m, odd: 32w+2m+1) so each lane's A-gather is one b32 covering
// both tiles' f16 values at the same column. B is tile-independent: built
// once per K-chunk, reused by both WMMAs. 64-edge chunks double-buffered in
// LDS via TDM (fallback: plain loads). 4 WMMAs per chunk, no tail masking.
// ---------------------------------------------------------------------------
__global__ void __launch_bounds__(256)
spmm_wmma_kernel(const int*          __restrict__ colBp,
                 const unsigned int* __restrict__ rwp,
                 const int*          __restrict__ tile_pstart,
                 const float*        __restrict__ z,
                 const _Float16*     __restrict__ pT,
                 float*              __restrict__ out) {
    __shared__ int      s_colB[2][64];
    __shared__ unsigned s_rw[2][64];

    const int tid     = threadIdx.x;
    const int wave    = tid >> 5;            // 0..7 : pair of batch tiles
    const int lane    = tid & 31;
    const int rowbase = blockIdx.x * 16;
    const int estart  = tile_pstart[blockIdx.x];
    const int nch     = (tile_pstart[blockIdx.x + 1] - estart) >> 6;  // exact

    const int m   = lane & 15;               // A: batch-pair-in-tile / B,D: node
    const int h   = lane >> 4;
    const unsigned um  = (unsigned)m;
    // byte offset of this lane's batch pair (batches 32w+2m, 32w+2m+1)
    const unsigned bgp = (unsigned)(wave * 32 + m * 2) * 2u;

    auto stage = [&](int buf, int base) {
#if USE_TDM
        if (wave == 0) {
            const unsigned rem = (base < PADMAX) ? (unsigned)(PADMAX - base) : 0u;
            tdm_load_64dw((unsigned)(uintptr_t)&s_colB[buf][0], colBp + base, rem);
            tdm_load_64dw((unsigned)(uintptr_t)&s_rw[buf][0],   rwp   + base, rem);
        }
#else
        if (tid < 64) {
            const int e  = base + tid;
            const int ec = (e < PADMAX) ? e : (PADMAX - 1);
            s_colB[buf][tid] = colBp[ec];
            s_rw[buf][tid]   = rwp[ec];
        }
#endif
    };

    v8f acc0 = {};   // even batch tile
    v8f acc1 = {};   // odd batch tile
    stage(0, estart);
    stage(1, estart + 64);

    for (int i = 0; i < nch; ++i) {
#if USE_TDM
        if (wave == 0) {
            if (i < nch - 1) __builtin_amdgcn_s_wait_tensorcnt(2); // chunk i landed
            else             __builtin_amdgcn_s_wait_tensorcnt(0); // drain
        }
#endif
        __syncthreads();                     // chunk i visible to all waves
        const int buf = i & 1;
#pragma unroll
        for (int q = 0; q < 2; ++q) {
            const int qq = q << 5;
            v16h a0, a1, b;
#pragma unroll
            for (int j = 0; j < 16; ++j) {
                // A 16-bit 16x32 layout: half j, lane-half h -> K = j + 8h + (j&8)
                const int ka = qq + j + (h << 3) + (j & 8);
                const unsigned boff = (unsigned)s_colB[buf][ka] + bgp;
                const unsigned pr = *(const unsigned*)((const char*)pT + boff);
                a0[j] = __builtin_bit_cast(_Float16, (unsigned short)(pr & 0xffffu));
                a1[j] = __builtin_bit_cast(_Float16, (unsigned short)(pr >> 16));
                // B 32x16 layout: same K order as A -> K = j + 8h + (j&8)
                const int kb = qq + j + (h << 3) + (j & 8);
                const unsigned v = s_rw[buf][kb];
                b[j] = ((v & 0xffffu) == um)
                           ? __builtin_bit_cast(_Float16, (unsigned short)(v >> 16))
                           : (_Float16)0.0f;
            }
            acc0 = __builtin_amdgcn_wmma_f32_16x16x32_f16(
                       false, a0, false, b, (short)0, acc0, false, false);
            acc1 = __builtin_amdgcn_wmma_f32_16x16x32_f16(
                       false, a1, false, b, (short)0, acc1, false, false);
        }
        __syncthreads();                     // everyone done with buf before rewrite
        if (i + 2 < nch) stage(buf, estart + ((i + 2) << 6));
    }

    // D layout: lane -> N = lane&15, VGPR g -> M = g + 8h
    // even tile: batch = 32w + 2M ; odd tile: batch = 32w + 2M + 1
    const int n = rowbase + m;
    const float zn = z[n];
#pragma unroll
    for (int g = 0; g < 8; ++g) {
        const int M = g + (h << 3);
        const int be = wave * 32 + 2 * M;
        out[(size_t)be * NN + n]       = __logf(acc0[g]) - zn;
        out[(size_t)(be + 1) * NN + n] = __logf(acc1[g]) - zn;
    }
}

// ---------------------------------------------------------------------------
extern "C" void kernel_launch(void* const* d_in, const int* in_sizes, int n_in,
                              void* d_out, int out_size, void* d_ws, size_t ws_size,
                              hipStream_t stream) {
    const float* child_ll = (const float*)d_in[0];   // [B, C]
    const float* log_w    = (const float*)d_in[1];   // [NSE]
    const int*   rows     = (const int*)d_in[2];     // [NSE]
    const int*   cols     = (const int*)d_in[3];     // [NSE]
    float*       out      = (float*)d_out;           // [B, N]

    char* ws = (char*)d_ws;
    size_t off = 0;
    _Float16* pT     = (_Float16*)(ws + off); off += (size_t)CC * BB * 2;   // 16 MB
    int* colBp       = (int*)(ws + off);      off += (size_t)PADMAX * 4;
    unsigned* rwp    = (unsigned*)(ws + off); off += (size_t)PADMAX * 4;
    int* row_start   = (int*)(ws + off);      off += (size_t)(NN + 1) * 4;
    int* tile_pstart = (int*)(ws + off);      off += (size_t)(NT + 1) * 4;
    float* z         = (float*)(ws + off);

    exp_transpose_kernel<<<dim3(CC / 32, BB / 32), dim3(32, 8), 0, stream>>>(child_ll, pT);
    build_row_start_kernel<<<NSE / 256, 256, 0, stream>>>(rows, row_start);
    tile_scan_kernel<<<1, NT, 0, stream>>>(row_start, tile_pstart);
    fill_pad_kernel<<<PADMAX / 256, 256, 0, stream>>>(colBp, rwp);
    scatter_edges_kernel<<<NSE / 256, 256, 0, stream>>>(log_w, rows, cols, row_start,
                                                        tile_pstart, colBp, rwp);
    compute_z_kernel<<<NN / 8, 256, 0, stream>>>(log_w, row_start, z);
    spmm_wmma_kernel<<<NT, 256, 0, stream>>>(colBp, rwp, tile_pstart, z, pT, out);
}
